// MDTA_FOR_VIDEO_49237505081643
// MI455X (gfx1250) — matrix operand order repaired and checked by reading, not compiled, not run on hardware
//
#include <hip/hip_runtime.h>
#include <math.h>

// ---------------- problem constants (from reference setup_inputs) -----------
#define BB    2
#define CC    128
#define C2    256
#define HH    128
#define WW    128
#define NPIX  (HH*WW)      // 16384
#define NHEAD 8
#define DH    16
#define GG    8

typedef __attribute__((ext_vector_type(16))) _Float16 v16h;
typedef __attribute__((ext_vector_type(8)))  float    v8f;

__device__ __forceinline__ v8f wmma16(v16h a, v16h b, v8f c) {
  // D = A(16x32 f16) * B(32x16 f16) + C(16x16 f32)
  return __builtin_amdgcn_wmma_f32_16x16x32_f16(false, a, false, b, (short)0, c,
                                                false, false);
}
__device__ __forceinline__ float sigmoidf(float v) {
  return 1.0f / (1.0f + __expf(-v));
}

// ---------------- weight packing (f32 -> f16, fragment-order) ---------------

// conv3x3 weights [Cout][Cin][9] -> wp[tap][n][ci] (n padded to Coutp, zeros)
__global__ void pack3_kernel(const float* __restrict__ w, _Float16* __restrict__ wp,
                             int Cout, int Coutp, int Cin) {
  int idx = blockIdx.x * 256 + threadIdx.x;
  int total = 9 * Coutp * Cin;
  if (idx >= total) return;
  int ci = idx % Cin;
  int n  = (idx / Cin) % Coutp;
  int tap = idx / (Cin * Coutp);
  float v = (n < Cout) ? w[((size_t)n * Cin + ci) * 9 + tap] : 0.0f;
  wp[idx] = (_Float16)v;
}

// conv1x1 weights: plain f32 -> f16 copy (layout already [n][k])
__global__ void pack1_kernel(const float* __restrict__ w, _Float16* __restrict__ wp,
                             int total) {
  int idx = blockIdx.x * 256 + threadIdx.x;
  if (idx >= total) return;
  wp[idx] = (_Float16)w[idx];
}

// ---------------- elementwise / data-movement kernels -----------------------

// xcat = concat([y, x], dim=1)  -> [B, 2C, H, W]
__global__ void concat_kernel(const float* __restrict__ x, const float* __restrict__ y,
                              float* __restrict__ xcat) {
  size_t idx = (size_t)blockIdx.x * blockDim.x + threadIdx.x;
  const size_t total = (size_t)BB * C2 * NPIX;
  if (idx >= total) return;
  int n = (int)(idx % NPIX);
  int c = (int)((idx / NPIX) % C2);
  int b = (int)(idx / ((size_t)C2 * NPIX));
  float v = (c < CC) ? y[((size_t)b * CC + c) * NPIX + n]
                     : x[((size_t)b * CC + (c - CC)) * NPIX + n];
  xcat[idx] = v;
}

// 2x2 average pool: [B,2C,128,128] -> [B,2C,64,64]
__global__ void avgpool_kernel(const float* __restrict__ in, float* __restrict__ out) {
  size_t idx = (size_t)blockIdx.x * blockDim.x + threadIdx.x;
  const size_t total = (size_t)BB * C2 * 64 * 64;
  if (idx >= total) return;
  int ox = (int)(idx % 64);
  int oy = (int)((idx / 64) % 64);
  int c  = (int)((idx / 4096) % C2);
  int b  = (int)(idx / ((size_t)C2 * 4096));
  const float* p = in + ((size_t)b * C2 + c) * NPIX;
  int y0 = oy * 2, x0 = ox * 2;
  out[idx] = 0.25f * (p[y0 * WW + x0] + p[y0 * WW + x0 + 1] +
                      p[(y0 + 1) * WW + x0] + p[(y0 + 1) * WW + x0 + 1]);
}

// t3 *= sigmoid(xcat + nearest_interp(a62 /*62x62*/ -> 128x128))
__global__ void gate_kernel(const float* __restrict__ xcat, const float* __restrict__ a62,
                            float* __restrict__ t3) {
  size_t idx = (size_t)blockIdx.x * blockDim.x + threadIdx.x;
  const size_t total = (size_t)BB * C2 * NPIX;
  if (idx >= total) return;
  int n = (int)(idx % NPIX);
  int c = (int)((idx / NPIX) % C2);
  int b = (int)(idx / ((size_t)C2 * NPIX));
  int yy = n >> 7, xx = n & 127;
  int iy = (yy * 62) >> 7, ix = (xx * 62) >> 7;   // == floor(y * 62/128)
  float av = a62[((size_t)b * C2 + c) * 3844 + iy * 62 + ix];
  t3[idx] *= sigmoidf(xcat[idx] + av);
}

// depthwise 3x3 pad=1 over Ct channels (branchless OOB)
__global__ void dw3x3_kernel(const float* __restrict__ in, const float* __restrict__ w,
                             float* __restrict__ out, int Bn, int Ct) {
  size_t idx = (size_t)blockIdx.x * blockDim.x + threadIdx.x;
  const size_t total = (size_t)Bn * Ct * NPIX;
  if (idx >= total) return;
  int n = (int)(idx % NPIX);
  int c = (int)((idx / NPIX) % Ct);
  const float* p = in + (idx - (size_t)n);  // plane base for this (b,c)
  const float* wc = w + (size_t)c * 9;
  int yy = n >> 7, xx = n & 127;
  float acc = 0.0f;
#pragma unroll
  for (int ky = 0; ky < 3; ++ky)
#pragma unroll
    for (int kx = 0; kx < 3; ++kx) {
      int iy = yy + ky - 1, ix = xx + kx - 1;
      bool ok = (unsigned)iy < (unsigned)HH && (unsigned)ix < (unsigned)WW;
      int iyc = ok ? iy : 0, ixc = ok ? ix : 0;
      float v = p[iyc * WW + ixc];               // always-safe load
      acc += (ok ? v : 0.0f) * wc[ky * 3 + kx];  // select, no branch
    }
  out[idx] = acc;
}

// ---------------- WMMA implicit-GEMM 3x3 convolution ------------------------
// out[b][n][m] = sum_{tap,ci} in[b][ci][oy+dy-pad][ox+dx-pad] * wp[tap][n][ci]
// K traversed tap-major (k' = tap*Cin + ci), same order for A and B.
// A: base + ci*chstride loads (immediate offsets).  B: packed f16, 16
// contiguous halves per fragment -> aligned v16h (2x global_load_b128).
// One wave computes a 16(M) x 64(N) tile; A fragment reused 4x.
template <bool MTAIL>
__global__ void conv3x3_wmma_kernel(const float* __restrict__ in,
                                    const _Float16* __restrict__ wp,
                                    float* __restrict__ out,
                                    int Bn, int Cin, int Hi, int Wi, int pad,
                                    int Cout, int Coutp, int Ho, int Wo) {
  const int M = Ho * Wo;
  const int mtiles = (M + 15) >> 4;
  const int nblocks = Coutp >> 6;
  const int total = Bn * mtiles * nblocks;
  int wave = blockIdx.x * (blockDim.x >> 5) + (threadIdx.x >> 5);
  if (wave >= total) return;             // wave-uniform: EXEC stays full below
  int b = wave / (mtiles * nblocks);
  int rem = wave - b * mtiles * nblocks;
  int mt = rem / nblocks;
  int nb = rem - mt * nblocks;
  int lane = threadIdx.x & 31;
  int half = lane >> 4;                  // 0: lanes 0-15, 1: lanes 16-31
  int r = lane & 15;
  int m0 = mt << 4;
  int m = m0 + r;
  bool mvalid = MTAIL ? (m < M) : true;
  int mm = mvalid ? m : 0;
  int oy = mm / Wo;
  int ox = mm - oy * Wo;
  const float* inb = in + (size_t)b * Cin * Hi * Wi;
  const size_t chs = (size_t)Hi * Wi;    // channel stride (floats)

  v8f acc[4] = {};
#pragma unroll
  for (int tap = 0; tap < 9; ++tap) {    // dy,dx become compile-time constants
    const int dy = tap / 3, dx = tap % 3;
    int iy = oy + dy - pad, ix = ox + dx - pad;
    bool aval = mvalid && (unsigned)iy < (unsigned)Hi && (unsigned)ix < (unsigned)Wi;
    const float* abase = inb + (aval ? ((size_t)iy * Wi + ix) : 0);
    // packed-B row bases for the 4 sub-tiles (f16, ci contiguous)
    const _Float16* wt = wp + (size_t)tap * Coutp * Cin;
    const _Float16* wb0 = wt + (size_t)((nb << 6) + 0  + r) * Cin;
    const _Float16* wb1 = wt + (size_t)((nb << 6) + 16 + r) * Cin;
    const _Float16* wb2 = wt + (size_t)((nb << 6) + 32 + r) * Cin;
    const _Float16* wb3 = wt + (size_t)((nb << 6) + 48 + r) * Cin;
    for (int c0 = 0; c0 < Cin; c0 += 32) {
      // A fragment: lane row = m; contiguous k order (same as B)
      //  lanes 0-15: halves 0-15 -> k c0+0..15
      //  lanes16-31: halves 0-15 -> k c0+16..31
      v16h a;
#pragma unroll
      for (int i = 0; i < 16; ++i) {
        int koff = half * 16 + i;
        float v = abase[(size_t)(c0 + koff) * chs];        // always-safe load
        a[i] = aval ? (_Float16)v : (_Float16)0.0f;        // branchless select
      }
      // B fragments: lane col = n; lanes 0-15 -> k c0+0..15, 16-31 -> c0+16..31
      int kb = c0 + half * 16;             // 32B-aligned (c0,half*16 mult of 16)
      v16h bf0 = *reinterpret_cast<const v16h*>(wb0 + kb);
      v16h bf1 = *reinterpret_cast<const v16h*>(wb1 + kb);
      v16h bf2 = *reinterpret_cast<const v16h*>(wb2 + kb);
      v16h bf3 = *reinterpret_cast<const v16h*>(wb3 + kb);
      acc[0] = wmma16(a, bf0, acc[0]);
      acc[1] = wmma16(a, bf1, acc[1]);
      acc[2] = wmma16(a, bf2, acc[2]);
      acc[3] = wmma16(a, bf3, acc[3]);
    }
  }
  // D layout: vgpr j -> (M = m0 + 8*half + j, N = n)
#pragma unroll
  for (int t = 0; t < 4; ++t) {
    int n = (nb << 6) + (t << 4) + r;
    if (n >= Cout) continue;             // only trims k4's padded channels
#pragma unroll
    for (int j = 0; j < 8; ++j) {
      int mrow = m0 + half * 8 + j;
      if (!MTAIL || mrow < M) out[((size_t)b * Cout + n) * M + mrow] = acc[t][j];
    }
  }
}

// ---------------- WMMA 1x1 convolution (batched GEMM) -----------------------
// out[b][n][m] = sum_k wp[n][k] * in[b][k][m]  (+ optional bias); wp is f16
__global__ void conv1x1_wmma_kernel(const float* __restrict__ in,
                                    const _Float16* __restrict__ wp,
                                    const float* __restrict__ bias,
                                    float* __restrict__ out,
                                    int Bn, int Cin, int Cout, int Np) {
  const int mtiles = Np >> 4;
  const int ntiles = Cout >> 4;
  const int total = Bn * mtiles * ntiles;
  int wave = blockIdx.x * (blockDim.x >> 5) + (threadIdx.x >> 5);
  if (wave >= total) return;             // wave-uniform
  int b = wave / (mtiles * ntiles);
  int rem = wave - b * mtiles * ntiles;
  int mt = rem / ntiles;
  int nt = rem - mt * ntiles;
  int lane = threadIdx.x & 31, half = lane >> 4, r = lane & 15;
  int m = (mt << 4) + r;
  int n = (nt << 4) + r;
  const float* abase = in + (size_t)b * Cin * Np + m;   // + k*Np (imm offsets)
  const _Float16* bbase = wp + (size_t)n * Cin;         // contiguous f16

  v8f acc = {};
  for (int k0 = 0; k0 < Cin; k0 += 32) {
    v16h a;
#pragma unroll
    for (int i = 0; i < 16; ++i) {
      int koff = half * 16 + i;
      a[i] = (_Float16)abase[(size_t)(k0 + koff) * Np];
    }
    v16h bf = *reinterpret_cast<const v16h*>(bbase + k0 + half * 16);
    acc = wmma16(a, bf, acc);
  }
  float bn = bias ? bias[n] : 0.0f;
#pragma unroll
  for (int j = 0; j < 8; ++j) {
    int mrow = (mt << 4) + half * 8 + j;
    out[((size_t)b * Cout + n) * Np + mrow] = acc[j] + bn;
  }
}

// ---------------- modulated deformable conv (k=3, pad=1, G=8) ---------------
__global__ void deform_kernel(const float* __restrict__ ybuf,
                              const float* __restrict__ off,
                              const float* __restrict__ dcn_w,
                              const float* __restrict__ dcn_b,
                              float* __restrict__ feat) {
  int pix = blockIdx.x;                 // b*NPIX + n
  int b = pix / NPIX;
  int n = pix - b * NPIX;
  int yy = n >> 7, xx = n & 127;
  int tid = threadIdx.x;                // 128 threads

  __shared__ float s[9][CC];
  __shared__ int   sy0[9], sx0[9];
  __shared__ float sfy[9], sfx[9], smk[9];

  if (tid < 9) {
    int tap = tid;
    const float* ob = off + (size_t)b * 18 * NPIX;
    float dy = ob[(size_t)(2 * tap) * NPIX + n];
    float dx = ob[(size_t)(2 * tap + 1) * NPIX + n];
    smk[tap] = sigmoidf(ob[(size_t)tap * NPIX + n]);     // mask = sigmoid(off[:9])
    float py = dy + (float)(yy - 1 + tap / 3);
    float px = dx + (float)(xx - 1 + (tap % 3));
    float fy0 = floorf(py), fx0 = floorf(px);
    sy0[tap] = (int)fy0; sx0[tap] = (int)fx0;
    sfy[tap] = py - fy0; sfx[tap] = px - fx0;
  }
  __syncthreads();

  const float* ch = ybuf + ((size_t)b * CC + tid) * NPIX;
#pragma unroll
  for (int tap = 0; tap < 9; ++tap) {
    int y0 = sy0[tap], x0 = sx0[tap];
    float fy = sfy[tap], fx = sfx[tap];
    float v = 0.0f;
#pragma unroll
    for (int cy = 0; cy < 2; ++cy)
#pragma unroll
      for (int cx = 0; cx < 2; ++cx) {
        int yi = y0 + cy, xi = x0 + cx;
        bool ok = (unsigned)yi < (unsigned)HH && (unsigned)xi < (unsigned)WW;
        int yc = ok ? yi : 0, xc = ok ? xi : 0;
        float wgt = (cy ? fy : 1.0f - fy) * (cx ? fx : 1.0f - fx);
        float pv = ch[yc * WW + xc];               // always-safe load
        v += pv * (ok ? wgt : 0.0f);               // select, no branch
      }
    s[tap][tid] = v * smk[tap];
  }
  __syncthreads();

  int o = tid;
  int g = o >> 4;
  const float* wo = dcn_w + (size_t)o * 16 * 9;          // [o][i][tap]
  float acc = dcn_b[o];
#pragma unroll
  for (int i = 0; i < 16; ++i) {
    int ci = (g << 4) + i;
#pragma unroll
    for (int tap = 0; tap < 9; ++tap)
      acc += s[tap][ci] * wo[i * 9 + tap];
  }
  feat[((size_t)b * CC + o) * NPIX + n] = fmaxf(acc, 0.0f);  // fused ReLU
}

// ---------------- l2 normalization over the pixel dimension -----------------
__global__ void l2norm_kernel(float* __restrict__ buf, int Cstride, int Cnorm) {
  int row = blockIdx.x;
  int b = row / Cnorm, c = row - b * Cnorm;
  float* p = buf + ((size_t)b * Cstride + c) * NPIX;
  __shared__ float red[256];
  float sacc = 0.0f;
  for (int i = threadIdx.x; i < NPIX; i += 256) {
    float v = p[i];
    sacc += v * v;
  }
  red[threadIdx.x] = sacc;
  __syncthreads();
  for (int st = 128; st > 0; st >>= 1) {
    if (threadIdx.x < st) red[threadIdx.x] += red[threadIdx.x + st];
    __syncthreads();
  }
  float inv = 1.0f / fmaxf(sqrtf(red[0]), 1e-12f);
  for (int i = threadIdx.x; i < NPIX; i += 256) p[i] *= inv;
}

// ---------------- attention: S = q k^T * temp, softmax over d ---------------
__global__ void attn_kernel(const float* __restrict__ q, const float* __restrict__ kv,
                            const float* __restrict__ temp, float* __restrict__ attn) {
  int blk = blockIdx.x;                 // b*128 + h*16 + c
  int c = blk & 15;
  int h = (blk >> 4) & 7;
  int b = blk >> 7;
  const float* qp = q + ((size_t)b * CC + h * DH + c) * NPIX;
  const float* kb = kv + ((size_t)b * C2 + h * DH) * NPIX;  // k = kv[:, :C]
  float part[16];
#pragma unroll
  for (int d = 0; d < 16; ++d) part[d] = 0.0f;
  for (int n = threadIdx.x; n < NPIX; n += 256) {
    float qv = qp[n];
#pragma unroll
    for (int d = 0; d < 16; ++d) part[d] += qv * kb[(size_t)d * NPIX + n];
  }
  __shared__ float wred[8][16];
  int lane = threadIdx.x & 31, wid = threadIdx.x >> 5;
#pragma unroll
  for (int d = 0; d < 16; ++d) {
    float v = part[d];
    for (int offl = 16; offl > 0; offl >>= 1) v += __shfl_down(v, offl, 32);
    if (lane == 0) wred[wid][d] = v;
  }
  __syncthreads();
  if (threadIdx.x == 0) {
    float t = temp[h];
    float sv[16];
    float mx = -1e30f;
#pragma unroll
    for (int d = 0; d < 16; ++d) {
      float v = 0.0f;
      for (int wdx = 0; wdx < 8; ++wdx) v += wred[wdx][d];
      v *= t;
      sv[d] = v;
      mx = fmaxf(mx, v);
    }
    float sum = 0.0f;
#pragma unroll
    for (int d = 0; d < 16; ++d) { sv[d] = __expf(sv[d] - mx); sum += sv[d]; }
    float inv = 1.0f / sum;
    float* ap = attn + (size_t)(b * 8 + h) * 256 + c * 16;
#pragma unroll
    for (int d = 0; d < 16; ++d) ap[d] = sv[d] * inv;
  }
}

// ---------------- out = attn @ v --------------------------------------------
__global__ void av_kernel(const float* __restrict__ attn, const float* __restrict__ kv,
                          float* __restrict__ av) {
  const int chunks = NPIX / 256;        // 64
  int blk = blockIdx.x;
  int chunk = blk % chunks;
  int h = (blk / chunks) % NHEAD;
  int b = blk / (chunks * NHEAD);
  __shared__ float at[16][16];
  at[threadIdx.x >> 4][threadIdx.x & 15] =
      attn[(size_t)(b * 8 + h) * 256 + threadIdx.x];
  __syncthreads();
  int n = chunk * 256 + threadIdx.x;
  const float* vb = kv + ((size_t)b * C2 + CC + h * DH) * NPIX;  // v = kv[:, C:]
  float vreg[16];
#pragma unroll
  for (int d = 0; d < 16; ++d) vreg[d] = vb[(size_t)d * NPIX + n];
#pragma unroll
  for (int c = 0; c < 16; ++c) {
    float acc = 0.0f;
#pragma unroll
    for (int d = 0; d < 16; ++d) acc += at[c][d] * vreg[d];
    av[((size_t)b * CC + h * DH + c) * NPIX + n] = acc;
  }
}

// ---------------- host orchestration ----------------------------------------
extern "C" void kernel_launch(void* const* d_in, const int* in_sizes, int n_in,
                              void* d_out, int out_size, void* d_ws, size_t ws_size,
                              hipStream_t stream) {
  (void)in_sizes; (void)n_in; (void)out_size; (void)ws_size;
  const float* x     = (const float*)d_in[0];
  const float* y     = (const float*)d_in[1];
  const float* q_w   = (const float*)d_in[2];
  const float* qd_w  = (const float*)d_in[3];
  const float* kv_w  = (const float*)d_in[4];
  const float* kvd_w = (const float*)d_in[5];
  const float* proj_w= (const float*)d_in[6];
  const float* temp  = (const float*)d_in[7];
  const float* k2_w  = (const float*)d_in[8];
  const float* k3_w  = (const float*)d_in[9];
  const float* k4_w  = (const float*)d_in[10];
  const float* dcn_w = (const float*)d_in[11];
  const float* dcn_b = (const float*)d_in[12];
  const float* pw_w  = (const float*)d_in[13];
  const float* pw_b  = (const float*)d_in[14];
  float* out = (float*)d_out;

  // workspace layout (floats), lifetime-based aliasing (~149 MiB total)
  float* ws      = (float*)d_ws;
  float* xcat    = ws;                         // [2,256,16384]  8388608
  float* t3      = xcat    + 8388608;          // [2,256,16384]  8388608
  float* feat    = t3      + 8388608;          // [2,128,16384]  4194304
  float* aligned = feat    + 4194304;          // [2,128,16384]  4194304
  float* qpre    = aligned + 4194304;          // [2,128,16384]  4194304
  float* qbuf    = qpre    + 4194304;          // [2,128,16384]  4194304
  float* pool    = qbuf    + 4194304;          // [2,256,64,64]  2097152
  float* a62     = pool    + 2097152;          // [2,256,62,62]  1968128
  float* offb    = a62     + 1968128;          // [2,18,16384]   589824
  float* attnb   = offb    + 589824;           // [2,8,16,16]    4096
  float* kvpre   = xcat;                       // alias: xcat dead after gate
  float* kvdw    = t3;                         // alias: t3 dead after k4
  float* avb     = feat;                       // alias: feat dead after pw
  // packed f16 weights (all offsets multiples of 16 halves -> 32B aligned)
  _Float16* pk2 = (_Float16*)(attnb + 4096);   // 9*256*256 = 589824 halves
  _Float16* pk3 = pk2 + 589824;                // 589824
  _Float16* pk4 = pk3 + 589824;                // 9*64*256 = 147456 (padded)
  _Float16* pq  = pk4 + 147456;                // 128*128 = 16384
  _Float16* pkv = pq  + 16384;                 // 256*128 = 32768
  _Float16* ppw = pkv + 32768;                 // 16384
  _Float16* ppj = ppw + 16384;                 // 16384

  auto c3 = [&](const float* in, const _Float16* wp, float* o,
                int Cin, int Hi, int Wi, int pad, int Cout, int Coutp) {
    int Ho = Hi + 2 * pad - 2, Wo = Wi + 2 * pad - 2;
    int M = Ho * Wo;
    int mtiles = (M + 15) / 16, nblocks = Coutp / 64;
    int waves = BB * mtiles * nblocks;
    int blocks = (waves + 7) / 8;
    if (M % 16)
      conv3x3_wmma_kernel<true><<<blocks, 256, 0, stream>>>(
          in, wp, o, BB, Cin, Hi, Wi, pad, Cout, Coutp, Ho, Wo);
    else
      conv3x3_wmma_kernel<false><<<blocks, 256, 0, stream>>>(
          in, wp, o, BB, Cin, Hi, Wi, pad, Cout, Coutp, Ho, Wo);
  };
  auto c1 = [&](const float* in, const _Float16* wp, const float* bias, float* o,
                int Cin, int Cout) {
    int waves = BB * (NPIX / 16) * (Cout / 16);
    conv1x1_wmma_kernel<<<(waves + 7) / 8, 256, 0, stream>>>(
        in, wp, bias, o, BB, Cin, Cout, NPIX);
  };

  // 0) pack weights to f16 fragment layouts (one cheap pass, L2-resident)
  pack3_kernel<<<(9 * 256 * 256 + 255) / 256, 256, 0, stream>>>(k2_w, pk2, 256, 256, 256);
  pack3_kernel<<<(9 * 256 * 256 + 255) / 256, 256, 0, stream>>>(k3_w, pk3, 256, 256, 256);
  pack3_kernel<<<(9 * 64 * 256 + 255) / 256, 256, 0, stream>>>(k4_w, pk4, 18, 64, 256);
  pack1_kernel<<<(128 * 128 + 255) / 256, 256, 0, stream>>>(q_w, pq, 128 * 128);
  pack1_kernel<<<(256 * 128 + 255) / 256, 256, 0, stream>>>(kv_w, pkv, 256 * 128);
  pack1_kernel<<<(128 * 128 + 255) / 256, 256, 0, stream>>>(pw_w, ppw, 128 * 128);
  pack1_kernel<<<(128 * 128 + 255) / 256, 256, 0, stream>>>(proj_w, ppj, 128 * 128);

  // 1) xcat = concat(y, x)
  concat_kernel<<<(BB * C2 * NPIX) / 256, 256, 0, stream>>>(x, y, xcat);
  // 2) SCConv: pooled -> k2 (pad 0) -> gate -> k3 (pad 1) * gate -> k4 -> offset
  avgpool_kernel<<<(BB * C2 * 64 * 64) / 256, 256, 0, stream>>>(xcat, pool);
  c3(pool, pk2, a62, C2, 64, 64, 0, C2, C2);              // 62x62 (M tail)
  c3(xcat, pk3, t3, C2, 128, 128, 1, C2, C2);
  gate_kernel<<<(BB * C2 * NPIX) / 256, 256, 0, stream>>>(xcat, a62, t3);
  c3(t3, pk4, offb, C2, 128, 128, 1, 18, 64);             // offsets (N padded)
  // 3) modulated deformable conv on y (+bias, ReLU fused)
  deform_kernel<<<BB * NPIX, 128, 0, stream>>>(y, offb, dcn_w, dcn_b, feat);
  // 4) aligned = pw(relu(feat)) + pw_b
  c1(feat, ppw, pw_b, aligned, CC, CC);
  // 5) q path
  c1(x, pq, nullptr, qpre, CC, CC);
  dw3x3_kernel<<<(BB * CC * NPIX) / 256, 256, 0, stream>>>(qpre, qd_w, qbuf, BB, CC);
  // 6) kv path on aligned
  c1(aligned, pkv, nullptr, kvpre, CC, C2);
  dw3x3_kernel<<<(BB * C2 * NPIX) / 256, 256, 0, stream>>>(kvpre, kvd_w, kvdw, BB, C2);
  // 7) l2norm q and k (k = first C channels of kvdw)
  l2norm_kernel<<<BB * CC, 256, 0, stream>>>(qbuf, CC, CC);
  l2norm_kernel<<<BB * CC, 256, 0, stream>>>(kvdw, C2, CC);
  // 8) attention
  attn_kernel<<<BB * CC, 256, 0, stream>>>(qbuf, kvdw, temp, attnb);
  av_kernel<<<BB * NHEAD * (NPIX / 256), 256, 0, stream>>>(attnb, kvdw, avb);
  // 9) projection -> output
  c1(avb, ppj, nullptr, out, CC, CC);
}
